// ClassifierRegressor_63324997812587
// MI455X (gfx1250) — matrix operand order repaired and checked by reading, not compiled, not run on hardware
//
#include <hip/hip_runtime.h>
#include <math.h>

// ---------------------------------------------------------------------------
// Fast R-CNN head for MI455X (gfx1250, wave32, WMMA).
//   GEMM1 (2048x25088x4096) bf16 WMMA w/ f32 accum, A-tile via
//   GLOBAL_LOAD_ASYNC_TO_LDS_B128 (ASYNCcnt), B-tile f32->bf16 in flight.
// ---------------------------------------------------------------------------

typedef __attribute__((ext_vector_type(16))) __bf16 v16bf;
typedef __attribute__((ext_vector_type(8)))  float  v8f;

#define GM 2048      // B*N rois
#define GN 4096      // hidden H
#define GK 25088     // C*Hp*Wp
#define MBLK 128
#define NBLK 128
#define KSTEP 32
#define LDSTR 20     // padded row stride in dwords (32 bf16 = 16 dwords + 4 pad)
#define NCLS 21
#define NB 512       // rois per batch

__device__ __forceinline__ unsigned pack2bf16(float a, float b) {
  unsigned ua = __float_as_uint(a), ub = __float_as_uint(b);
  ua = (ua + 0x7FFFu + ((ua >> 16) & 1u)) >> 16;   // RNE truncate
  ub = (ub + 0x7FFFu + ((ub >> 16) & 1u)) >> 16;
  return (ub << 16) | (ua & 0xFFFFu);
}

// ---- kernel 0: feats f32 -> bf16 (one L2-resident copy reused by 32 N-panels)
__global__ void __launch_bounds__(256)
cvt_feats_bf16(const float* __restrict__ rois, unsigned* __restrict__ abf) {
  size_t i = (size_t)blockIdx.x * 256 + threadIdx.x;   // one uint4 = 8 elems
  const float4* s = (const float4*)rois;
  float4 f0 = s[2 * i], f1 = s[2 * i + 1];
  uint4 o;
  o.x = pack2bf16(f0.x, f0.y); o.y = pack2bf16(f0.z, f0.w);
  o.z = pack2bf16(f1.x, f1.y); o.w = pack2bf16(f1.z, f1.w);
  ((uint4*)abf)[i] = o;
}

// ---- kernel 1: x = feats * W1^T + b1, bf16 WMMA, 128x128 block, K-step 32
__global__ void __launch_bounds__(256)
gemm1_wmma(const unsigned* __restrict__ Abf, const float* __restrict__ W1,
           const float* __restrict__ b1, float* __restrict__ X) {
  __shared__ unsigned ldsA[2][MBLK * LDSTR];
  __shared__ unsigned ldsB[2][MBLK * LDSTR];

  const int tid  = threadIdx.x;
  const int lane = tid & 31, wid = tid >> 5;
  const int wm = wid & 1, wn = wid >> 1;        // wave tile: 64 rows x 32 cols
  const int ml = lane & 15, hf = lane >> 4;
  const int m0 = blockIdx.x * MBLK;             // M fast -> W1 panel L2 reuse
  const int n0 = blockIdx.y * NBLK;
  const int ar = tid >> 1, ah = tid & 1;        // staging: row 0..127, half 0/1

  v8f acc[4][2] = {};

  // per-buffer LDS byte addresses for the async copies (low 32 bits of the
  // generic pointer are the LDS offset on CDNA5)
  const unsigned ldsAoff[2] = {
      (unsigned)(size_t)&ldsA[0][ar * LDSTR + ah * 8],
      (unsigned)(size_t)&ldsA[1][ar * LDSTR + ah * 8]};
  const unsigned* const arow = Abf + (((size_t)(m0 + ar) * GK) >> 1) + ah * 8;

  // A tile: bf16 copy straight into LDS via the async engine (ASYNCcnt)
  auto stageA = [&](int buf, int k0) {
    const unsigned* gsrc = arow + (k0 >> 1);
    asm volatile("global_load_async_to_lds_b128 %0, %1, off"
                 :: "v"(ldsAoff[buf]), "v"(gsrc) : "memory");
  };

  // B tile: W1 row n, f32 -> bf16 converted in flight, 128 x 32
  auto stageB = [&](int buf, int k0) {
    const float4* gb = (const float4*)(W1 + (size_t)(n0 + ar) * GK + k0 + ah * 16);
    float4 f0 = gb[0], f1 = gb[1], f2 = gb[2], f3 = gb[3];
    if (k0 + 2 * KSTEP < GK)   // prefetch next-next W1 slab (global_prefetch_b8)
      __builtin_prefetch(W1 + (size_t)(n0 + ar) * GK + (k0 + 2 * KSTEP) + ah * 16, 0, 1);
    unsigned* db = &ldsB[buf][ar * LDSTR + ah * 8];
    db[0] = pack2bf16(f0.x, f0.y); db[1] = pack2bf16(f0.z, f0.w);
    db[2] = pack2bf16(f1.x, f1.y); db[3] = pack2bf16(f1.z, f1.w);
    db[4] = pack2bf16(f2.x, f2.y); db[5] = pack2bf16(f2.z, f2.w);
    db[6] = pack2bf16(f3.x, f3.y); db[7] = pack2bf16(f3.z, f3.w);
  };

  union Frag { uint4 q[2]; v16bf v; };

  auto compute = [&](int buf) {
    Frag a[4], b[2];
    // B frag (K = 16*hf + 2v + j, contiguous): dwords [8hf .. 8hf+7] of row n
    #pragma unroll
    for (int u = 0; u < 2; ++u) {
      const unsigned* p = &ldsB[buf][(wn * 32 + u * 16 + ml) * LDSTR + hf * 8];
      b[u].q[0] = *(const uint4*)(p);
      b[u].q[1] = *(const uint4*)(p + 4);
    }
    // A frag (K = 16*hf + 2v + j, contiguous): dwords [8hf .. 8hf+7] of row m
    #pragma unroll
    for (int t = 0; t < 4; ++t) {
      const unsigned* p = &ldsA[buf][(wm * 64 + t * 16 + ml) * LDSTR];
      a[t].q[0] = *(const uint4*)(p + 8 * hf);
      a[t].q[1] = *(const uint4*)(p + 8 * hf + 4);
    }
    #pragma unroll
    for (int t = 0; t < 4; ++t)
      #pragma unroll
      for (int u = 0; u < 2; ++u)
        acc[t][u] = __builtin_amdgcn_wmma_f32_16x16x32_bf16(
            false, a[t].v, false, b[u].v, (short)0, acc[t][u], false, false);
  };

  stageA(0, 0);
  stageB(0, 0);
  asm volatile("s_wait_asynccnt 0x0" ::: "memory");
  __syncthreads();
  const int nk = GK / KSTEP;   // 784
  for (int ks = 0; ks < nk; ++ks) {
    int cur = ks & 1;
    if (ks + 1 < nk) {
      stageA(cur ^ 1, (ks + 1) * KSTEP);   // DMA overlaps the WMMAs below
      stageB(cur ^ 1, (ks + 1) * KSTEP);
    }
    compute(cur);
    asm volatile("s_wait_asynccnt 0x0" ::: "memory");
    __syncthreads();
  }

  // epilogue: C/D layout  (m = r + 8*hf, n = ml)
  #pragma unroll
  for (int t = 0; t < 4; ++t) {
    int mrow = m0 + wm * 64 + t * 16 + hf * 8;
    #pragma unroll
    for (int u = 0; u < 2; ++u) {
      int ncol = n0 + wn * 32 + u * 16 + ml;
      float bias = b1[ncol];
      float* o = X + (size_t)mrow * GN + ncol;
      #pragma unroll
      for (int r = 0; r < 8; ++r)
        o[(size_t)r * GN] = acc[t][u][r] + bias;
    }
  }
}

// ---- kernel 2: per-roi 25 dot products + softmax/argmax + box decode
__global__ void __launch_bounds__(256)
head_kernel(const float* __restrict__ X,
            const float* __restrict__ Wc, const float* __restrict__ bc,
            const float* __restrict__ Wr, const float* __restrict__ br,
            const float* __restrict__ proposals,
            float* __restrict__ score, float* __restrict__ boxes) {
  const int row = blockIdx.x;            // 0..2047  (b*512 + n)
  const int tid = threadIdx.x;
  const int lane = tid & 31, wid = tid >> 5;
  const float4* xp = (const float4*)(X + (size_t)row * GN + tid * 16);
  float4 xv[4];
  #pragma unroll
  for (int i = 0; i < 4; ++i) xv[i] = xp[i];

  __shared__ float wpart[8];
  __shared__ float logits[25];

  for (int j = 0; j < 25; ++j) {
    const float* wrow = (j < NCLS) ? (Wc + (size_t)j * GN) : (Wr + (size_t)(j - NCLS) * GN);
    const float4* wp = (const float4*)(wrow + tid * 16);
    float a = 0.f;
    #pragma unroll
    for (int i = 0; i < 4; ++i) {
      float4 w = wp[i];
      a += xv[i].x * w.x + xv[i].y * w.y + xv[i].z * w.z + xv[i].w * w.w;
    }
    #pragma unroll
    for (int off = 16; off; off >>= 1) a += __shfl_down(a, off, 32);
    if (lane == 0) wpart[wid] = a;
    __syncthreads();
    if (tid == 0) {
      float s = 0.f;
      #pragma unroll
      for (int w = 0; w < 8; ++w) s += wpart[w];
      logits[j] = s + ((j < NCLS) ? bc[j] : br[j - NCLS]);
    }
    __syncthreads();
  }

  if (tid == 0) {
    float lmax = logits[0]; int cls = 0;
    for (int c = 1; c < NCLS; ++c)
      if (logits[c] > lmax) { lmax = logits[c]; cls = c; }
    float sum = 0.f;
    for (int c = 0; c < NCLS; ++c) sum += expf(logits[c] - lmax);
    float sc = 1.f / sum;                        // max softmax prob
    bool valid = (cls != 0) && (sc >= 0.01f);
    const float* p = proposals + (size_t)row * 4;
    float px = p[0] + p[2] * logits[21];
    float py = p[1] + p[3] * logits[22];
    float pw = p[2] * expf(logits[23]);
    float ph = p[3] * expf(logits[24]);
    score[row] = valid ? sc : 0.f;
    float* bo = boxes + (size_t)row * 4;
    bo[0] = px; bo[1] = py; bo[2] = pw; bo[3] = ph;
  }
}

// ---- kernel 3: per-batch NMS (stable sort by -score, greedy suppression)
__global__ void __launch_bounds__(256)
nms_kernel(const float* __restrict__ score, const float* __restrict__ boxes,
           float* __restrict__ out) {
  const int b = blockIdx.x, tid = threadIdx.x;
  __shared__ float sc[NB]; __shared__ int id[NB];
  __shared__ float x0s[NB], y0s[NB], x1s[NB], y1s[NB], ar[NB];
  __shared__ int vld[NB], sup[NB];

  for (int i = tid; i < NB; i += 256) { sc[i] = score[b * NB + i]; id[i] = i; }
  __syncthreads();

  // bitonic sort; key order: descending score, ascending original index (stable)
  for (int k = 2; k <= NB; k <<= 1) {
    for (int j = k >> 1; j > 0; j >>= 1) {
      for (int i = tid; i < NB; i += 256) {
        int ixj = i ^ j;
        if (ixj > i) {
          float s1 = sc[i], s2 = sc[ixj]; int d1 = id[i], d2 = id[ixj];
          bool less12 = (s1 > s2) || (s1 == s2 && d1 < d2);  // key[i] < key[ixj]
          bool doswap = ((i & k) == 0) ? (!less12) : less12;
          if (doswap) { sc[i] = s2; sc[ixj] = s1; id[i] = d2; id[ixj] = d1; }
        }
      }
      __syncthreads();
    }
  }

  for (int i = tid; i < NB; i += 256) {
    const float* bp = boxes + (size_t)(b * NB + id[i]) * 4;
    float px = bp[0], py = bp[1], pw = bp[2], ph = bp[3];
    float cx0 = fminf(fmaxf(px, 0.f), 599.f);
    float cy0 = fminf(fmaxf(py, 0.f), 599.f);
    float cx1 = fminf(fmaxf(px + pw - 1.f, 0.f), 599.f);
    float cy1 = fminf(fmaxf(py + ph - 1.f, 0.f), 599.f);
    x0s[i] = cx0; y0s[i] = cy0; x1s[i] = cx1; y1s[i] = cy1;
    ar[i] = fmaxf(cx1 - cx0 + 1.f, 0.f) * fmaxf(cy1 - cy0 + 1.f, 0.f);
    vld[i] = sc[i] > 0.f;
    sup[i] = 0;
  }
  __syncthreads();

  for (int i = 0; i < NB; ++i) {
    bool alive = (!sup[i]) && vld[i];
    if (alive) {
      float ix0 = x0s[i], iy0 = y0s[i], ix1 = x1s[i], iy1 = y1s[i], ia = ar[i];
      for (int j = i + 1 + tid; j < NB; j += 256) {
        float inter = fmaxf(fminf(ix1, x1s[j]) - fmaxf(ix0, x0s[j]) + 1.f, 0.f) *
                      fmaxf(fminf(iy1, y1s[j]) - fmaxf(iy0, y0s[j]) + 1.f, 0.f);
        float iou = inter / (ia + ar[j] - inter + 1e-9f);
        if (iou > 0.5f) sup[j] = 1;
      }
    }
    __syncthreads();
  }

  float* out_cls = out;          // (B,N)   = 2048 floats
  float* out_box = out + GM;     // (B,N,4) = 8192 floats
  for (int i = tid; i < NB; i += 256) {
    int d = id[i];
    bool keep = (!sup[i]) && vld[i];
    out_cls[b * NB + d] = keep ? sc[i] : 0.f;
    const float* bp = boxes + (size_t)(b * NB + d) * 4;
    float px = keep ? bp[0] : 0.f, py = keep ? bp[1] : 0.f;
    float pw = keep ? bp[2] : 0.f, ph = keep ? bp[3] : 0.f;
    float* ob = out_box + (size_t)(b * NB + d) * 4;
    ob[0] = px; ob[1] = py; ob[2] = px + pw - 1.f; ob[3] = py + ph - 1.f;
  }
}

extern "C" void kernel_launch(void* const* d_in, const int* in_sizes, int n_in,
                              void* d_out, int out_size, void* d_ws, size_t ws_size,
                              hipStream_t stream) {
  const float* rois      = (const float*)d_in[0];  // (4,512,512,7,7)
  const float* proposals = (const float*)d_in[1];  // (4,512,4)
  const float* W1        = (const float*)d_in[2];  // (4096,25088)
  const float* b1        = (const float*)d_in[3];  // (4096,)
  const float* Wc        = (const float*)d_in[4];  // (21,4096)
  const float* bc        = (const float*)d_in[5];  // (21,)
  const float* Wr        = (const float*)d_in[6];  // (4,4096)
  const float* br        = (const float*)d_in[7];  // (4,)

  char* w = (char*)d_ws;
  unsigned* Abf = (unsigned*)w;                                    // 102,760,448 B
  float* X   = (float*)(w + 102760448);                            //  33,554,432 B
  float* SCp = (float*)(w + 102760448 + 33554432);                 //       8,192 B
  float* BXp = (float*)(w + 102760448 + 33554432 + 8192);          //      32,768 B
  float* out = (float*)d_out;

  cvt_feats_bf16<<<25088, 256, 0, stream>>>(rois, Abf);            // 51,380,224 elems
  gemm1_wmma<<<dim3(GM / MBLK, GN / NBLK), 256, 0, stream>>>(Abf, W1, b1, X);
  head_kernel<<<GM, 256, 0, stream>>>(X, Wc, bc, Wr, br, proposals, SCp, BXp);
  nms_kernel<<<4, 256, 0, stream>>>(SCp, BXp, out);
}
